// VisionTextAttentionNoMask_25649544692240
// MI455X (gfx1250) — compile-verified, matrix operand order checked by reading, not run on hardware
//
#include <hip/hip_runtime.h>
#include <hip/hip_bf16.h>

typedef __bf16 bf16_t;
typedef __attribute__((ext_vector_type(16))) __bf16 v16bf;
typedef __attribute__((ext_vector_type(8)))  float  v8f;

__device__ __forceinline__ v8f vzero8() {
    v8f r;
#pragma unroll
    for (int i = 0; i < 8; ++i) r[i] = 0.f;
    return r;
}

// Load a 16-element bf16 WMMA operand fragment (A-layout, or B from an
// [n][k]-transposed LDS tile). rowp points at k-index 0 of this lane's row.
// koff already includes the lane-group offset: (lane&16)?8:0 (+ k-tile base).
// Element j<8  -> k = koff + j        (K 0..7   / 8..15)
// Element j>=8 -> k = koff + 16 + j-8 (K 16..23 / 24..31)
__device__ __forceinline__ v16bf ld_frag(const bf16_t* __restrict__ rowp, int koff) {
    v16bf f;
#pragma unroll
    for (int j = 0; j < 8; ++j) f[j] = rowp[koff + j];
#pragma unroll
    for (int j = 0; j < 8; ++j) f[8 + j] = rowp[koff + 16 + j];
    return f;
}

// ---------------------------------------------------------------------------
// Elementwise conversion kernels
// ---------------------------------------------------------------------------
__global__ __launch_bounds__(256) void cvt_f32_bf16(const float* __restrict__ in,
                                                    bf16_t* __restrict__ out, long n) {
    long i = (long)blockIdx.x * 256 + threadIdx.x;
    long stride = (long)gridDim.x * 256;
    for (long p = i; p < n; p += stride) out[p] = (bf16_t)in[p];
}

// out[c*R + r] = in[r*C + c]
__global__ __launch_bounds__(256) void transpose_f32_bf16(const float* __restrict__ in,
                                                          bf16_t* __restrict__ out,
                                                          int R, int C) {
    long i = (long)blockIdx.x * 256 + threadIdx.x;
    if (i < (long)R * C) {
        int r = (int)(i / C), c = (int)(i % C);
        out[(long)c * R + r] = (bf16_t)in[i];
    }
}

// ---------------------------------------------------------------------------
// Generic bf16 WMMA GEMM: C[M,N] = A[M,K] * B[K,N] + bias
// Block tile 128x256, BK=64. 256 threads = 8 waves (2 rows x 4 cols),
// each wave computes 64x64 via 4x4 v_wmma_f32_16x16x32_bf16 accumulators.
// ---------------------------------------------------------------------------
template <bool OUT_BF16, bool BIAS_ROW>
__global__ __launch_bounds__(256) void gemm_tile(
    const bf16_t* __restrict__ A, long long sAb, int lda,
    const bf16_t* __restrict__ B, long long sBb, int ldb,
    void* __restrict__ Cv, long long sCb, int ldc,
    const float* __restrict__ bias, int Kd) {
    __shared__ __align__(16) bf16_t sA[128 * 72];  // [m][k], padded
    __shared__ __align__(16) bf16_t sB[256 * 72];  // transposed: [n][k], padded

    const int bz = blockIdx.z;
    A += (long long)bz * sAb;
    B += (long long)bz * sBb;
    const long row0 = (long)blockIdx.y * 128;
    const int  col0 = blockIdx.x * 256;
    const int tid = threadIdx.x;
    const int w = tid >> 5, lane = tid & 31;
    const int lm = lane & 15;
    const int kg = (lane & 16) ? 8 : 0;
    const int wr = w >> 2, wc = w & 3;  // wave covers rows wr*64.., cols wc*64..

    v8f acc[4][4];
#pragma unroll
    for (int mi = 0; mi < 4; ++mi)
#pragma unroll
        for (int nj = 0; nj < 4; ++nj) acc[mi][nj] = vzero8();

    for (int k0 = 0; k0 < Kd; k0 += 64) {
        __syncthreads();
        if (k0 + 64 < Kd) {  // prefetch next k-tile -> global_prefetch_b8
            __builtin_prefetch(A + (row0 + (tid >> 1)) * (long)lda + k0 + 64, 0, 1);
            __builtin_prefetch(B + (long)(k0 + 64 + (tid >> 3)) * ldb + col0 +
                                   (tid & 7) * 32, 0, 1);
        }
        // Stage A tile [128][64] with 16B vector loads
#pragma unroll
        for (int g = tid; g < 1024; g += 256) {
            int r = g >> 3, c8 = (g & 7) * 8;
            uint4 u = *(const uint4*)(A + (row0 + r) * (long)lda + k0 + c8);
            *(uint4*)&sA[r * 72 + c8] = u;
        }
        // Stage B tile [64][256] transposed into sB[n][k]
#pragma unroll
        for (int g = tid; g < 2048; g += 256) {
            int kk = g >> 5, c8 = (g & 31) * 8;
            uint4 u = *(const uint4*)(B + (long)(k0 + kk) * ldb + col0 + c8);
            const bf16_t* sp = (const bf16_t*)&u;
#pragma unroll
            for (int i = 0; i < 8; ++i) sB[(c8 + i) * 72 + kk] = sp[i];
        }
        __syncthreads();

#pragma unroll
        for (int ks = 0; ks < 2; ++ks) {
            const int ko = ks * 32 + kg;
            v16bf afr[4], bfr[4];
#pragma unroll
            for (int mi = 0; mi < 4; ++mi)
                afr[mi] = ld_frag(sA + (wr * 64 + mi * 16 + lm) * 72, ko);
#pragma unroll
            for (int nj = 0; nj < 4; ++nj)
                bfr[nj] = ld_frag(sB + (wc * 64 + nj * 16 + lm) * 72, ko);
#pragma unroll
            for (int mi = 0; mi < 4; ++mi)
#pragma unroll
                for (int nj = 0; nj < 4; ++nj)
                    acc[mi][nj] = __builtin_amdgcn_wmma_f32_16x16x32_bf16(
                        false, afr[mi], false, bfr[nj], (short)0, acc[mi][nj],
                        false, false);
        }
    }

    // Epilogue
#pragma unroll
    for (int mi = 0; mi < 4; ++mi)
#pragma unroll
        for (int nj = 0; nj < 4; ++nj)
#pragma unroll
            for (int i = 0; i < 8; ++i) {
                long r = row0 + wr * 64 + mi * 16 + i + kg;  // kg==8 -> M rows 8..15
                int  c = col0 + wc * 64 + nj * 16 + lm;
                float bval = BIAS_ROW ? bias[(int)r] : bias[c];
                float val = acc[mi][nj][i] + bval;
                long long idx = (long long)bz * sCb + r * (long long)ldc + c;
                if (OUT_BF16) ((bf16_t*)Cv)[idx] = (bf16_t)val;
                else          ((float*)Cv)[idx]  = val;
            }
}

// ---------------------------------------------------------------------------
// GroupNorm stats (two-pass, deterministic). x: [16][4096][1024].
// ---------------------------------------------------------------------------
__global__ __launch_bounds__(256) void gn_partial(const float* __restrict__ x,
                                                  float* __restrict__ ps,
                                                  float* __restrict__ ps2) {
    int b = blockIdx.x, ch = blockIdx.y;  // 16 x 16, 256 rows per chunk
    const float* base = x + ((long)b * 4096 + (long)ch * 256) * 1024;
#pragma unroll
    for (int c4 = 0; c4 < 4; ++c4) {
        int c = c4 * 256 + threadIdx.x;
        float s = 0.f, s2 = 0.f;
        for (int r = 0; r < 256; ++r) {
            float v = base[(long)r * 1024 + c];
            s += v;
            s2 += v * v;
        }
        long o = ((long)b * 16 + ch) * 1024 + c;
        ps[o] = s;
        ps2[o] = s2;
    }
}

__global__ __launch_bounds__(256) void gn_finalize(const float* __restrict__ ps,
                                                   const float* __restrict__ ps2,
                                                   float* __restrict__ mean,
                                                   float* __restrict__ inv) {
    int idx = blockIdx.x * 256 + threadIdx.x;  // 64 blocks -> 16384 (b,c) pairs
    int b = idx >> 10, c = idx & 1023;
    float s = 0.f, s2 = 0.f;
#pragma unroll
    for (int ch = 0; ch < 16; ++ch) {
        long o = ((long)b * 16 + ch) * 1024 + c;
        s += ps[o];
        s2 += ps2[o];
    }
    float m = s * (1.f / 4096.f);
    float var = s2 * (1.f / 4096.f) - m * m;
    mean[idx] = m;
    inv[idx] = rsqrtf(var + 1e-5f);
}

__global__ __launch_bounds__(256) void gn_apply(float* __restrict__ x,
                                                const float* __restrict__ mean,
                                                const float* __restrict__ inv,
                                                const float* __restrict__ gamma,
                                                const float* __restrict__ beta) {
    const long total = 16L * 4096 * 1024;
    long i = (long)blockIdx.x * 256 + threadIdx.x;
    long stride = (long)gridDim.x * 256;
    for (long p = i; p < total; p += stride) {
        int c = (int)(p & 1023);
        int b = (int)(p >> 22);  // p / (4096*1024)
        int s = b * 1024 + c;
        x[p] = (x[p] - mean[s]) * inv[s] * gamma[c] + beta[c];
    }
}

// ---------------------------------------------------------------------------
// Fused attention: per (b,h) and 128 q-rows per workgroup.
//   qn = GN(q)*K^-0.5 (folded), sim = qn @ k, softmax_T, out = attn @ v^T
// Each of 8 waves owns 16 q-rows and ALL 256 sim columns in registers.
// ---------------------------------------------------------------------------
__global__ __launch_bounds__(256) void attn_kernel(
    const float* __restrict__ qraw,                    // [16*4096][1024]
    const float* __restrict__ qmean, const float* __restrict__ qinv,  // [16*1024]
    const float* __restrict__ gamma_q, const float* __restrict__ beta_q,
    const bf16_t* __restrict__ kmat,                   // [16][1024][256]
    const bf16_t* __restrict__ vmat,                   // [16][1024][256]
    bf16_t* __restrict__ outmat) {                     // [16*4096][1024]
    const int bh = blockIdx.y;
    const int b = bh >> 3, h = bh & 7;
    const int n0 = blockIdx.x * 128;
    const int tid = threadIdx.x;
    const int w = tid >> 5, lane = tid & 31;
    const int lm = lane & 15;
    const int hi8 = (lane & 16) ? 8 : 0;

    __shared__ __align__(16) bf16_t s_q[128 * 132];  // q tile (reused as attn slabs)
    __shared__ __align__(16) bf16_t s_kv[9216];      // k^T / v chunk

    // Stage + normalize q tile -> bf16 (fold GroupNorm and 1/sqrt(K))
    const float scale = 0.03125f;  // 1024^-0.5
    for (int i = tid; i < 128 * 128; i += 256) {
        int r = i >> 7, c = i & 127;
        int gc = h * 128 + c;
        float xv = qraw[((long)b * 4096 + n0 + r) * 1024 + gc];
        int si = b * 1024 + gc;
        float qn = ((xv - qmean[si]) * qinv[si] * gamma_q[gc] + beta_q[gc]) * scale;
        s_q[r * 132 + c] = (bf16_t)qn;
    }

    // ---- GEMM1: sim[16 rows][256 cols] per wave, T chunked by 64 ----
    v8f sim[16];
#pragma unroll
    for (int i = 0; i < 16; ++i) sim[i] = vzero8();

    for (int tc = 0; tc < 4; ++tc) {
        __syncthreads();
        // stage k^T chunk: s_kv[t_local][132] = k[b][h*128+d][tc*64+t]
        for (int g = tid; g < 1024; g += 256) {
            int d = g >> 3, t8 = (g & 7) * 8;
            uint4 u = *(const uint4*)(kmat +
                (((long)b * 1024 + h * 128 + d) * 256 + tc * 64 + t8));
            const bf16_t* sp = (const bf16_t*)&u;
#pragma unroll
            for (int i = 0; i < 8; ++i) s_kv[(t8 + i) * 132 + d] = sp[i];
        }
        __syncthreads();
#pragma unroll
        for (int ks = 0; ks < 4; ++ks) {  // d = ks*32 ..
            v16bf a = ld_frag(s_q + (w * 16 + lm) * 132, ks * 32 + hi8);
#pragma unroll
            for (int nl = 0; nl < 4; ++nl) {
                v16bf bb = ld_frag(s_kv + (nl * 16 + lm) * 132, ks * 32 + hi8);
                sim[tc * 4 + nl] = __builtin_amdgcn_wmma_f32_16x16x32_bf16(
                    false, a, false, bb, (short)0, sim[tc * 4 + nl], false, false);
            }
        }
    }

    // ---- softmax over 256 cols: rows live in (vgpr i, lane-group hi) ----
    float rmax[8], rsum[8];
#pragma unroll
    for (int i = 0; i < 8; ++i) {
        float m = -3.0e38f;
#pragma unroll
        for (int ni = 0; ni < 16; ++ni) m = fmaxf(m, sim[ni][i]);
        rmax[i] = m;
    }
#pragma unroll
    for (int mask = 1; mask < 16; mask <<= 1)
#pragma unroll
        for (int i = 0; i < 8; ++i)
            rmax[i] = fmaxf(rmax[i], __shfl_xor(rmax[i], mask, 32));
#pragma unroll
    for (int i = 0; i < 8; ++i) {
        float s = 0.f;
#pragma unroll
        for (int ni = 0; ni < 16; ++ni) {
            float e = __expf(sim[ni][i] - rmax[i]);
            sim[ni][i] = e;
            s += e;
        }
        rsum[i] = s;
    }
#pragma unroll
    for (int mask = 1; mask < 16; mask <<= 1)
#pragma unroll
        for (int i = 0; i < 8; ++i) rsum[i] += __shfl_xor(rsum[i], mask, 32);
#pragma unroll
    for (int i = 0; i < 8; ++i) {
        float r = 1.f / rsum[i];
#pragma unroll
        for (int ni = 0; ni < 16; ++ni) sim[ni][i] *= r;
    }

    // ---- GEMM2: out[16 rows][128 d] = attn @ v^T, T chunked by 64 ----
    v8f oacc[8];
#pragma unroll
    for (int i = 0; i < 8; ++i) oacc[i] = vzero8();
    bf16_t* slab = s_q + w * (16 * 40);  // wave-private attn A-fragment slab

    for (int tc = 0; tc < 4; ++tc) {
        __syncthreads();
        // stage v chunk: s_kv[d][72] = v[b][h*128+d][tc*64 + t]
        for (int g = tid; g < 1024; g += 256) {
            int d = g >> 3, t8 = (g & 7) * 8;
            uint4 u = *(const uint4*)(vmat +
                (((long)b * 1024 + h * 128 + d) * 256 + tc * 64 + t8));
            *(uint4*)&s_kv[d * 72 + t8] = u;
        }
        __syncthreads();
#pragma unroll
        for (int kk = 0; kk < 2; ++kk) {
            int g = tc * 2 + kk;  // attn cols g*32..g*32+31 -> sim tiles 2g,2g+1
#pragma unroll
            for (int half = 0; half < 2; ++half) {
                int ni = 2 * g + half;
#pragma unroll
                for (int i = 0; i < 8; ++i)
                    slab[(i + hi8) * 40 + half * 16 + lm] = (bf16_t)sim[ni][i];
            }
            v16bf a = ld_frag(slab + lm * 40, hi8);
#pragma unroll
            for (int nd = 0; nd < 8; ++nd) {
                v16bf bb = ld_frag(s_kv + (nd * 16 + lm) * 72, kk * 32 + hi8);
                oacc[nd] = __builtin_amdgcn_wmma_f32_16x16x32_bf16(
                    false, a, false, bb, (short)0, oacc[nd], false, false);
            }
        }
    }

    // Epilogue -> attn output [b][n][h*128 + d], bf16
#pragma unroll
    for (int nd = 0; nd < 8; ++nd)
#pragma unroll
        for (int i = 0; i < 8; ++i) {
            long r = (long)b * 4096 + n0 + w * 16 + i + hi8;
            int  c = h * 128 + nd * 16 + lm;
            outmat[r * 1024 + c] = (bf16_t)oacc[nd][i];
        }
}

// ---------------------------------------------------------------------------
extern "C" void kernel_launch(void* const* d_in, const int* in_sizes, int n_in,
                              void* d_out, int out_size, void* d_ws, size_t ws_size,
                              hipStream_t stream) {
    const float* x   = (const float*)d_in[0];
    const float* l   = (const float*)d_in[1];
    const float* Wq  = (const float*)d_in[2];
    const float* bq  = (const float*)d_in[3];
    const float* gq  = (const float*)d_in[4];
    const float* btq = (const float*)d_in[5];
    const float* Wk  = (const float*)d_in[6];
    const float* bk  = (const float*)d_in[7];
    const float* Wv  = (const float*)d_in[8];
    const float* bv  = (const float*)d_in[9];
    const float* Wo  = (const float*)d_in[10];
    const float* bo  = (const float*)d_in[11];
    const float* go  = (const float*)d_in[12];
    const float* bto = (const float*)d_in[13];
    float* out = (float*)d_out;

    char* ws = (char*)d_ws;
    size_t off = 0;
    auto take = [&](size_t bytes) -> void* {
        void* p = ws + off;
        off = (off + bytes + 255) & ~(size_t)255;
        return p;
    };
    bf16_t* xb   = (bf16_t*)take(65536L * 1024 * 2);
    bf16_t* lb   = (bf16_t*)take(16L * 768 * 256 * 2);
    bf16_t* wqT  = (bf16_t*)take(1024L * 1024 * 2);
    bf16_t* wkb  = (bf16_t*)take(1024L * 768 * 2);
    bf16_t* wvb  = (bf16_t*)take(1024L * 768 * 2);
    bf16_t* woT  = (bf16_t*)take(1024L * 1024 * 2);
    float*  qf   = (float*)take(65536L * 1024 * 4);
    bf16_t* kb   = (bf16_t*)take(16L * 1024 * 256 * 2);
    bf16_t* vb   = (bf16_t*)take(16L * 1024 * 256 * 2);
    bf16_t* ao   = (bf16_t*)take(65536L * 1024 * 2);
    float*  qmean = (float*)take(16384 * 4);
    float*  qinv  = (float*)take(16384 * 4);
    float*  omean = (float*)take(16384 * 4);
    float*  oinv  = (float*)take(16384 * 4);
    float*  ps    = (float*)take(262144 * 4);
    float*  ps2   = (float*)take(262144 * 4);

    // 1. precision conversion / weight transposition
    cvt_f32_bf16<<<8192, 256, 0, stream>>>(x, xb, 65536L * 1024);
    cvt_f32_bf16<<<2048, 256, 0, stream>>>(l, lb, 16L * 768 * 256);
    transpose_f32_bf16<<<4096, 256, 0, stream>>>(Wq, wqT, 1024, 1024);
    cvt_f32_bf16<<<1024, 256, 0, stream>>>(Wk, wkb, 1024L * 768);
    cvt_f32_bf16<<<1024, 256, 0, stream>>>(Wv, wvb, 1024L * 768);
    transpose_f32_bf16<<<4096, 256, 0, stream>>>(Wo, woT, 1024, 1024);

    // 2. Q projection (f32 out) + GroupNorm stats
    gemm_tile<false, false><<<dim3(4, 512, 1), 256, 0, stream>>>(
        xb, 0, 1024, wqT, 0, 1024, qf, 0, 1024, bq, 1024);
    gn_partial<<<dim3(16, 16), 256, 0, stream>>>(qf, ps, ps2);
    gn_finalize<<<64, 256, 0, stream>>>(ps, ps2, qmean, qinv);

    // 3. K/V projections (bf16 out, row-indexed bias), batched over grid.z
    gemm_tile<true, true><<<dim3(1, 8, 16), 256, 0, stream>>>(
        wkb, 0, 768, lb, 768L * 256, 256, kb, 1024L * 256, 256, bk, 768);
    gemm_tile<true, true><<<dim3(1, 8, 16), 256, 0, stream>>>(
        wvb, 0, 768, lb, 768L * 256, 256, vb, 1024L * 256, 256, bv, 768);

    // 4. fused GN(q) + attention + softmax
    attn_kernel<<<dim3(32, 128), 256, 0, stream>>>(qf, qmean, qinv, gq, btq,
                                                   kb, vb, ao);

    // 5. output projection into d_out, then GroupNorm in place
    gemm_tile<false, false><<<dim3(4, 512, 1), 256, 0, stream>>>(
        ao, 0, 1024, woT, 0, 1024, out, 0, 1024, bo, 1024);
    gn_partial<<<dim3(16, 16), 256, 0, stream>>>(out, ps, ps2);
    gn_finalize<<<64, 256, 0, stream>>>(ps, ps2, omean, oinv);
    gn_apply<<<8192, 256, 0, stream>>>(out, omean, oinv, go, bto);

    (void)in_sizes; (void)n_in; (void)out_size; (void)ws_size;
}
